// Seq2Seq_65266323030608
// MI455X (gfx1250) — matrix operand order repaired and checked by reading, not compiled, not run on hardware
//
#include <hip/hip_runtime.h>
#include <stdint.h>

// ---------------- problem constants ----------------
constexpr int kB = 32, kS = 128, kT = 64, kV = 32000, kE = 300, kH = 256;
constexpr int kEP = 320;               // E padded to multiple of 32
constexpr int kKE = kEP + kH;          // 576 : encoder packed K  [x | h]
constexpr int kKD = kEP + kH + kH;     // 832 : decoder packed K  [x | ctx | h]

// ---------------- WMMA plumbing (CDNA5 wave32) ----------------
typedef __attribute__((ext_vector_type(16))) __bf16 v16bf;
typedef __attribute__((ext_vector_type(8)))  float  v8f;

union Frag { uint4 u[2]; v16bf v; };

__device__ __forceinline__ v8f v8zero() {
  v8f z = {0.f,0.f,0.f,0.f,0.f,0.f,0.f,0.f};
  return z;
}

__device__ __forceinline__ unsigned short f2bf(float x) {
  union { float f; uint32_t u; } c; c.f = x;
  uint32_t r = c.u + 0x7FFFu + ((c.u >> 16) & 1u);   // round-nearest-even
  return (unsigned short)(r >> 16);
}
__device__ __forceinline__ float bfl(uint32_t u) {
  union { uint32_t u; float f; } c; c.u = u << 16; return c.f;
}
__device__ __forceinline__ float bfh(uint32_t u) {
  union { uint32_t u; float f; } c; c.u = u & 0xFFFF0000u; return c.f;
}
__device__ __forceinline__ float sigm(float x) { return 1.0f / (1.0f + __expf(-x)); }

__device__ __forceinline__ v8f wmma_bf(const Frag& a, const Frag& b, v8f c) {
  // D = A(16x32 bf16) * B(32x16 bf16) + C(16x16 f32)
  return __builtin_amdgcn_wmma_f32_16x16x32_bf16(false, a.v, false, b.v,
                                                 (short)0, c, false, false);
}

// A fragment: lane L (L&15)=row; lanes<16 hold K {k0..k0+7, k0+16..k0+23},
// lanes>=16 hold K {k0+8..k0+15, k0+24..k0+31}  (ISA 16-bit A 16x32 layout)
__device__ __forceinline__ void loadFragA(Frag& f, const unsigned short* rowBase,
                                          int k0, int lane) {
  int kk = k0 + ((lane & 16) ? 8 : 0);
  const uint4* p = (const uint4*)rowBase;
  f.u[0] = p[kk >> 3];
  f.u[1] = p[(kk >> 3) + 2];
}
// B fragment: B(k,n) = W[n,k]; lane reads W[n16+(lane&15), k0+((lane>>4)<<3)+{0..7,16..23}]
__device__ __forceinline__ void loadFragB(Frag& f, const unsigned short* W,
                                          int strideK, int n16, int k0, int lane) {
  const uint4* p = (const uint4*)(W + (size_t)(n16 + (lane & 15)) * strideK);
  int c = (k0 + ((lane >> 4) << 3)) >> 3;
  f.u[0] = p[c];
  f.u[1] = p[c + 2];
}

// ---------------- preprocessing kernels ----------------
__global__ __launch_bounds__(256) void cvt_bf16_kernel(const float* __restrict__ in,
                                                       unsigned short* __restrict__ out,
                                                       int n) {
  for (int i = blockIdx.x * blockDim.x + threadIdx.x; i < n; i += gridDim.x * blockDim.x)
    out[i] = f2bf(in[i]);
}

// encW packed: (1024 x 576) = [Wih(:,0:300) | pad20 | Whh(:,0:256)]
__global__ __launch_bounds__(256) void pack_enc_kernel(const float* __restrict__ Wih,
                                                       const float* __restrict__ Whh,
                                                       unsigned short* __restrict__ out) {
  int n = 4 * kH * kKE;
  for (int i = blockIdx.x * blockDim.x + threadIdx.x; i < n; i += gridDim.x * blockDim.x) {
    int g = i / kKE, c = i % kKE;
    float v = 0.f;
    if (c < kE)        v = Wih[(size_t)g * kE + c];
    else if (c >= kEP) v = Whh[(size_t)g * kH + (c - kEP)];
    out[i] = f2bf(v);
  }
}
// decW packed: (1024 x 832) = [Wih(:,0:300) | pad20 | Wih(:,300:556) | Whh]
__global__ __launch_bounds__(256) void pack_dec_kernel(const float* __restrict__ Wih,
                                                       const float* __restrict__ Whh,
                                                       unsigned short* __restrict__ out) {
  int n = 4 * kH * kKD;
  const int ld = kE + kH;  // 556
  for (int i = blockIdx.x * blockDim.x + threadIdx.x; i < n; i += gridDim.x * blockDim.x) {
    int g = i / kKD, c = i % kKD;
    float v = 0.f;
    if (c < kE)        v = Wih[(size_t)g * ld + c];
    else if (c < kEP)  v = 0.f;
    else if (c < kKE)  v = Wih[(size_t)g * ld + kE + (c - kEP)];
    else               v = Whh[(size_t)g * kH + (c - kKE)];
    out[i] = f2bf(v);
  }
}

// embeddings gathered time-major: out[(l*B + b)*EP + e]
__global__ __launch_bounds__(256) void gather_embed_kernel(const int* __restrict__ tok,
                                                           const float* __restrict__ emb,
                                                           unsigned short* __restrict__ out,
                                                           int L) {
  int n = L * kB * kEP;
  for (int i = blockIdx.x * blockDim.x + threadIdx.x; i < n; i += gridDim.x * blockDim.x) {
    int e = i % kEP;
    int r = i / kEP;
    int b = r % kB, l = r / kB;
    float v = 0.f;
    if (e < kE) v = emb[(size_t)tok[b * L + l] * kE + e];
    out[i] = f2bf(v);
  }
}

// ---------------- encoder: 1 persistent workgroup, 16 waves ----------------
__global__ __launch_bounds__(512) void encoder_kernel(
    const unsigned short* __restrict__ srcE,   // (S, B, EP) bf16
    const unsigned short* __restrict__ encW,   // (1024, 576) bf16 packed
    const float* __restrict__ bih, const float* __restrict__ bhh,
    float* __restrict__ encOut,                // (B, S, H) f32
    unsigned short* __restrict__ encOutB,      // (B, S, H) bf16
    float* __restrict__ cState)                // (B, H) f32 scratch
{
  __shared__ unsigned short hbf[kB * kH];      // 16KB: h in bf16 (A-fragment source)
  const int tid = threadIdx.x, lane = tid & 31, wave = tid >> 5;

  for (int i = tid; i < kB * kH; i += 512) { hbf[i] = 0; cState[i] = 0.f; }
  __syncthreads();

#pragma unroll 1
  for (int t = 0; t < kS; ++t) {
    // Opaque barrier: weights are const/restrict; without this LICM hoists all
    // t-invariant B-fragments into registers -> massive scratch spills.
    // L2 (192MB) holds the 1.1MB packed weights, so re-reading per step is cheap.
    asm volatile("" ::: "memory");

    v8f acc[2][4];
#pragma unroll
    for (int ti = 0; ti < 2; ++ti)
#pragma unroll
      for (int g = 0; g < 4; ++g) acc[ti][g] = v8zero();

#pragma unroll
    for (int ti = 0; ti < 2; ++ti) {
      const int task = wave + 16 * ti;             // 0..31
      const int mhalf = task >> 4, ngroup = task & 15;
      const int arow = mhalf * 16 + (lane & 15);
      const unsigned short* aX = srcE + (size_t)(t * kB + arow) * kEP;
      const unsigned short* aH = hbf + arow * kH;
      // branch-free K segments: x (global) then h (LDS)
#pragma unroll 1
      for (int ks = 0; ks < kEP / 32; ++ks) {
        Frag a; loadFragA(a, aX, ks * 32, lane);
#pragma unroll
        for (int g = 0; g < 4; ++g) {              // gate order i,f,g,o
          Frag bb;
          loadFragB(bb, encW, kKE, (g * 16 + ngroup) * 16, ks * 32, lane);
          acc[ti][g] = wmma_bf(a, bb, acc[ti][g]);
        }
      }
#pragma unroll 1
      for (int ks = 0; ks < kH / 32; ++ks) {
        Frag a; loadFragA(a, aH, ks * 32, lane);
#pragma unroll
        for (int g = 0; g < 4; ++g) {
          Frag bb;
          loadFragB(bb, encW, kKE, (g * 16 + ngroup) * 16, kEP + ks * 32, lane);
          acc[ti][g] = wmma_bf(a, bb, acc[ti][g]);
        }
      }
    }
    __syncthreads();  // all reads of old h complete before anyone writes new h

#pragma unroll
    for (int ti = 0; ti < 2; ++ti) {
      const int task = wave + 16 * ti;
      const int mhalf = task >> 4, ngroup = task & 15;
      const int u = ngroup * 16 + (lane & 15);
      const float b_i = bih[0 * kH + u] + bhh[0 * kH + u];
      const float b_f = bih[1 * kH + u] + bhh[1 * kH + u];
      const float b_g = bih[2 * kH + u] + bhh[2 * kH + u];
      const float b_o = bih[3 * kH + u] + bhh[3 * kH + u];
#pragma unroll
      for (int r = 0; r < 8; ++r) {
        const int brow = mhalf * 16 + r + ((lane & 16) ? 8 : 0);
        const float gi = acc[ti][0][r] + b_i;
        const float gf = acc[ti][1][r] + b_f;
        const float gg = acc[ti][2][r] + b_g;
        const float go = acc[ti][3][r] + b_o;
        const int ci = brow * kH + u;
        float c = sigm(gf) * cState[ci] + sigm(gi) * tanhf(gg);
        float h = sigm(go) * tanhf(c);
        cState[ci] = c;
        hbf[ci] = f2bf(h);
        const size_t oi = (size_t)(brow * kS + t) * kH + u;
        encOut[oi]  = h;
        encOutB[oi] = f2bf(h);
      }
    }
    __syncthreads();  // new h visible for next step
  }
}

// ---------------- decoder: 1 persistent workgroup, 16 waves ----------------
__global__ __launch_bounds__(512) void decoder_kernel(
    const unsigned short* __restrict__ tgtE,    // (T, B, EP) bf16
    const unsigned short* __restrict__ decW,    // (1024, 832) bf16 packed
    const float* __restrict__ bih, const float* __restrict__ bhh,
    const float* __restrict__ alignedA,         // (B, S, H) f32
    const float* __restrict__ encOut,           // (B, S, H) f32
    const int* __restrict__ srcMask,            // (B, S)
    float* __restrict__ cState,                 // (B, H) f32 (carried from encoder)
    unsigned short* __restrict__ Zb)            // (T*B, 512) bf16 : [h | ctx]
{
  __shared__ unsigned short hbf[kB * kH];       // 16KB
  __shared__ unsigned short ctxb[kB * kH];      // 16KB
  __shared__ float wAtt[kB * kS];               // 16KB
  const int tid = threadIdx.x, lane = tid & 31, wave = tid >> 5;

  // h0 = last encoder hidden state; c carried in cState
  for (int i = tid; i < kB * kH; i += 512) {
    int b = i >> 8, u = i & (kH - 1);
    hbf[i] = f2bf(encOut[(size_t)(b * kS + (kS - 1)) * kH + u]);
  }
  __syncthreads();

#pragma unroll 1
  for (int t = 0; t < kT; ++t) {
    asm volatile("" ::: "memory");   // defeat cross-step hoist of weight loads

    // ---- scores[b,s] = aligned[b,s,:] . h[b,:]  (f32 VALU, bf16 h) ----
#pragma unroll
    for (int it = 0; it < 8; ++it) {
      const int p = tid + 512 * it;              // 0..4095
      const int b = p >> 7, s = p & (kS - 1);
      const float4* ar = (const float4*)(alignedA + ((size_t)b * kS + s) * kH);
      const uint4*  hr = (const uint4*)(hbf + b * kH);
      float acc = 0.f;
#pragma unroll 1
      for (int k8 = 0; k8 < kH / 8; ++k8) {
        uint4 hv = hr[k8];
        float4 a0 = ar[2 * k8], a1 = ar[2 * k8 + 1];
        acc = fmaf(a0.x, bfl(hv.x), acc); acc = fmaf(a0.y, bfh(hv.x), acc);
        acc = fmaf(a0.z, bfl(hv.y), acc); acc = fmaf(a0.w, bfh(hv.y), acc);
        acc = fmaf(a1.x, bfl(hv.z), acc); acc = fmaf(a1.y, bfh(hv.z), acc);
        acc = fmaf(a1.z, bfl(hv.w), acc); acc = fmaf(a1.w, bfh(hv.w), acc);
      }
      if (srcMask[b * kS + s] == 0) acc = -1e10f;
      wAtt[p] = acc;
    }
    __syncthreads();

    // ---- softmax over s: half-wave (16 lanes) per batch row ----
    {
      const int b = wave * 2 + (lane >> 4);
      const int l16 = lane & 15;
      float e[kS / 16];
      float m = -3.4e38f;
#pragma unroll
      for (int j = 0; j < kS / 16; ++j) { e[j] = wAtt[b * kS + l16 + j * 16]; m = fmaxf(m, e[j]); }
#pragma unroll
      for (int d = 1; d < 16; d <<= 1) m = fmaxf(m, __shfl_xor(m, d, 16));
      float sum = 0.f;
#pragma unroll
      for (int j = 0; j < kS / 16; ++j) { e[j] = __expf(e[j] - m); sum += e[j]; }
#pragma unroll
      for (int d = 1; d < 16; d <<= 1) sum += __shfl_xor(sum, d, 16);
      const float inv = 1.f / sum;
#pragma unroll
      for (int j = 0; j < kS / 16; ++j) wAtt[b * kS + l16 + j * 16] = e[j] * inv;
    }
    __syncthreads();

    // ---- context[b,u] = sum_s w[b,s] * encOut[b,s,u] ----
#pragma unroll
    for (int it = 0; it < 16; ++it) {
      const int o = tid + 512 * it;              // 0..8191
      const int b = o >> 8, u = o & (kH - 1);
      float acc = 0.f;
#pragma unroll 1
      for (int s = 0; s < kS; ++s)
        acc = fmaf(wAtt[b * kS + s], encOut[(size_t)(b * kS + s) * kH + u], acc);
      const unsigned short cb = f2bf(acc);
      ctxb[o] = cb;
      Zb[(size_t)(t * kB + b) * 512 + kH + u] = cb;   // ctx half of z_t
    }
    __syncthreads();

    // ---- gate GEMM: gates(32x1024) = [x|ctx|h](32x832) @ decW^T ----
    v8f acc[2][4];
#pragma unroll
    for (int ti = 0; ti < 2; ++ti)
#pragma unroll
      for (int g = 0; g < 4; ++g) acc[ti][g] = v8zero();

#pragma unroll
    for (int ti = 0; ti < 2; ++ti) {
      const int task = wave + 16 * ti;
      const int mhalf = task >> 4, ngroup = task & 15;
      const int arow = mhalf * 16 + (lane & 15);
      const unsigned short* aX = tgtE + (size_t)(t * kB + arow) * kEP;
      const unsigned short* aC = ctxb + arow * kH;
      const unsigned short* aH = hbf + arow * kH;
      // branch-free K segments: x (global), ctx (LDS), h (LDS)
#pragma unroll 1
      for (int ks = 0; ks < kEP / 32; ++ks) {
        Frag a; loadFragA(a, aX, ks * 32, lane);
#pragma unroll
        for (int g = 0; g < 4; ++g) {
          Frag bb;
          loadFragB(bb, decW, kKD, (g * 16 + ngroup) * 16, ks * 32, lane);
          acc[ti][g] = wmma_bf(a, bb, acc[ti][g]);
        }
      }
#pragma unroll 1
      for (int ks = 0; ks < kH / 32; ++ks) {
        Frag a; loadFragA(a, aC, ks * 32, lane);
#pragma unroll
        for (int g = 0; g < 4; ++g) {
          Frag bb;
          loadFragB(bb, decW, kKD, (g * 16 + ngroup) * 16, kEP + ks * 32, lane);
          acc[ti][g] = wmma_bf(a, bb, acc[ti][g]);
        }
      }
#pragma unroll 1
      for (int ks = 0; ks < kH / 32; ++ks) {
        Frag a; loadFragA(a, aH, ks * 32, lane);
#pragma unroll
        for (int g = 0; g < 4; ++g) {
          Frag bb;
          loadFragB(bb, decW, kKD, (g * 16 + ngroup) * 16, kKE + ks * 32, lane);
          acc[ti][g] = wmma_bf(a, bb, acc[ti][g]);
        }
      }
    }
    __syncthreads();

    // ---- LSTM elementwise update, store z_t h-half ----
#pragma unroll
    for (int ti = 0; ti < 2; ++ti) {
      const int task = wave + 16 * ti;
      const int mhalf = task >> 4, ngroup = task & 15;
      const int u = ngroup * 16 + (lane & 15);
      const float b_i = bih[0 * kH + u] + bhh[0 * kH + u];
      const float b_f = bih[1 * kH + u] + bhh[1 * kH + u];
      const float b_g = bih[2 * kH + u] + bhh[2 * kH + u];
      const float b_o = bih[3 * kH + u] + bhh[3 * kH + u];
#pragma unroll
      for (int r = 0; r < 8; ++r) {
        const int brow = mhalf * 16 + r + ((lane & 16) ? 8 : 0);
        const float gi = acc[ti][0][r] + b_i;
        const float gf = acc[ti][1][r] + b_f;
        const float gg = acc[ti][2][r] + b_g;
        const float go = acc[ti][3][r] + b_o;
        const int ci = brow * kH + u;
        float c = sigm(gf) * cState[ci] + sigm(gi) * tanhf(gg);
        float h = sigm(go) * tanhf(c);
        cState[ci] = c;
        hbf[ci] = f2bf(h);
        Zb[(size_t)(t * kB + brow) * 512 + u] = f2bf(h);
      }
    }
    __syncthreads();
  }
}

// ---------------- generic bf16 WMMA GEMM: C(MxN) = A(MxK) @ W(NxK)^T + bias ----------------
// block: 256 threads (8 waves); each block computes a 128x64 tile.
// W block (64 rows x K) staged in LDS once (K fully resident).
template <int K, bool REMAP>
__global__ __launch_bounds__(256) void gemm_bf16_kernel(
    const unsigned short* __restrict__ A,   // (M, K) bf16 row-major
    const unsigned short* __restrict__ W,   // (N, K) bf16 row-major
    const float* __restrict__ bias,         // length N or nullptr
    float* __restrict__ out, int Nout)
{
  __shared__ unsigned short Bsh[64 * K];    // 64KB (K=512) / 32KB (K=256)
  const int tid = threadIdx.x, lane = tid & 31, wave = tid >> 5;
  const int nb = blockIdx.x, mb = blockIdx.y;

  const uint4* Wg = (const uint4*)(W + (size_t)nb * 64 * K);
  uint4* Bs4 = (uint4*)Bsh;
#pragma unroll 1
  for (int i = tid; i < 64 * K / 8; i += 256) Bs4[i] = Wg[i];

  const int m0 = mb * 128 + wave * 16;
  const unsigned short* Arow = A + (size_t)(m0 + (lane & 15)) * K;
  __builtin_prefetch(Arow, 0, 0);           // -> global_prefetch_b8
  __syncthreads();

  v8f acc[4];
#pragma unroll
  for (int nt = 0; nt < 4; ++nt) acc[nt] = v8zero();

  // unroll 1: per-iteration live set = 1 A frag + 4 B frags + accs -> no spills;
  // latency hidden by 8 resident waves, not by giant unrolled windows.
#pragma unroll 1
  for (int ks = 0; ks < K / 32; ++ks) {
    Frag a;
    loadFragA(a, Arow, ks * 32, lane);
#pragma unroll
    for (int nt = 0; nt < 4; ++nt) {
      Frag bb;
      loadFragB(bb, Bsh, K, nt * 16, ks * 32, lane);
      acc[nt] = wmma_bf(a, bb, acc[nt]);
    }
  }

#pragma unroll
  for (int nt = 0; nt < 4; ++nt) {
    const int col = nb * 64 + nt * 16 + (lane & 15);
    const float bs = bias ? bias[col] : 0.f;
#pragma unroll
    for (int r = 0; r < 8; ++r) {
      const int grow = m0 + r + ((lane & 16) ? 8 : 0);
      size_t oi;
      if (REMAP) {                            // row = t*B + b  ->  out[b, t, col]
        const int b = grow & (kB - 1), t = grow >> 5;
        oi = ((size_t)b * kT + t) * kV + col;
      } else {
        oi = (size_t)grow * Nout + col;
      }
      out[oi] = acc[nt][r] + bs;
    }
  }
}

// ---------------- host launch ----------------
extern "C" void kernel_launch(void* const* d_in, const int* in_sizes, int n_in,
                              void* d_out, int out_size, void* d_ws, size_t ws_size,
                              hipStream_t stream) {
  (void)in_sizes; (void)n_in; (void)out_size; (void)ws_size;
  const int*   src     = (const int*)d_in[0];
  const int*   tgt     = (const int*)d_in[1];
  const int*   srcMask = (const int*)d_in[2];
  const float* srcEmb  = (const float*)d_in[3];
  const float* tgtEmb  = (const float*)d_in[4];
  const float* encWih  = (const float*)d_in[5];
  const float* encWhh  = (const float*)d_in[6];
  const float* encBih  = (const float*)d_in[7];
  const float* encBhh  = (const float*)d_in[8];
  const float* attnW   = (const float*)d_in[9];
  const float* decWih  = (const float*)d_in[10];
  const float* decWhh  = (const float*)d_in[11];
  const float* decBih  = (const float*)d_in[12];
  const float* decBhh  = (const float*)d_in[13];
  const float* fcW     = (const float*)d_in[14];
  const float* fcB     = (const float*)d_in[15];

  // workspace carve-out (~52.4 MB total), 256B-aligned slices
  char* ws = (char*)d_ws;
  size_t off = 0;
  auto carve = [&](size_t bytes) { char* p = ws + off; off += (bytes + 255) & ~(size_t)255; return p; };
  unsigned short* fcWb    = (unsigned short*)carve((size_t)kV * 512 * 2);       // 32.77 MB
  unsigned short* encWb   = (unsigned short*)carve((size_t)4 * kH * kKE * 2);   // 1.18 MB
  unsigned short* decWb   = (unsigned short*)carve((size_t)4 * kH * kKD * 2);   // 1.70 MB
  unsigned short* attnWb  = (unsigned short*)carve((size_t)kH * kH * 2);        // 0.13 MB
  unsigned short* srcE    = (unsigned short*)carve((size_t)kS * kB * kEP * 2);  // 2.62 MB
  unsigned short* tgtE    = (unsigned short*)carve((size_t)kT * kB * kEP * 2);  // 1.31 MB
  float*          encOut  = (float*)carve((size_t)kB * kS * kH * 4);            // 4.19 MB
  unsigned short* encOutB = (unsigned short*)carve((size_t)kB * kS * kH * 2);   // 2.10 MB
  float*          alignedB= (float*)carve((size_t)kB * kS * kH * 4);            // 4.19 MB
  unsigned short* Zb      = (unsigned short*)carve((size_t)kT * kB * 512 * 2);  // 2.10 MB
  float*          cState  = (float*)carve((size_t)kB * kH * 4);                 // 32 KB

  // --- phase 0: precision conversion / packing / embedding gather ---
  cvt_bf16_kernel<<<8192, 256, 0, stream>>>(fcW, fcWb, kV * 512);
  cvt_bf16_kernel<<<256, 256, 0, stream>>>(attnW, attnWb, kH * kH);
  pack_enc_kernel<<<2304, 256, 0, stream>>>(encWih, encWhh, encWb);
  pack_dec_kernel<<<3328, 256, 0, stream>>>(decWih, decWhh, decWb);
  gather_embed_kernel<<<5120, 256, 0, stream>>>(src, srcEmb, srcE, kS);
  gather_embed_kernel<<<2560, 256, 0, stream>>>(tgt, tgtEmb, tgtE, kT);

  // --- phase 1: encoder recurrence (latency-bound; 1 persistent WGP) ---
  encoder_kernel<<<1, 512, 0, stream>>>(srcE, encWb, encBih, encBhh,
                                        encOut, encOutB, cState);

  // --- phase 2: aligned = enc_out @ attn_W^T  (4096x256x256 WMMA GEMM) ---
  gemm_bf16_kernel<kH, false><<<dim3(kH / 64, (kB * kS) / 128), 256, 0, stream>>>(
      encOutB, attnWb, (const float*)nullptr, alignedB, kH);

  // --- phase 3: decoder recurrence (attention + LSTM; z_t saved for later) ---
  decoder_kernel<<<1, 512, 0, stream>>>(tgtE, decWb, decBih, decBhh,
                                        alignedB, encOut, srcMask, cState, Zb);

  // --- phase 4: logits = Z @ fc_W^T + fc_b  (2048x32000x512, HBM-store-bound) ---
  gemm_bf16_kernel<512, true><<<dim3(kV / 64, (kT * kB) / 128), 256, 0, stream>>>(
      Zb, fcWb, fcB, (float*)d_out, kV);
}
